// MoEMLP_61890478735806
// MI455X (gfx1250) — compile-verified, matrix operand order checked by reading, not run on hardware
//
#include <hip/hip_runtime.h>
#include <math.h>

// Problem constants (match reference)
#define TT 8192      // B*S tokens
#define HH 2048      // hidden
#define II 5632      // intermediate
#define EE 8         // experts
#define KKTOP 2      // top-k
#define SLOT_CAP 16896   // T*K (16384) + per-expert 64-alignment padding (<=504)
#define MAXTILES 264     // worst case sum_e ceil(count_e/64) <= T*K/64 + E

typedef __attribute__((ext_vector_type(16))) __bf16      v16bf;
typedef __attribute__((ext_vector_type(8)))  float       v8f;
typedef __attribute__((ext_vector_type(4)))  unsigned int v4u;

__device__ __forceinline__ unsigned short f2bf(float f) {
    unsigned int u = __float_as_uint(f);
    u += 0x7FFFu + ((u >> 16) & 1u);          // round-to-nearest-even
    return (unsigned short)(u >> 16);
}

// Load a 16x(K=32) bf16 fragment from LDS laid out with 32-bf16 row stride.
// Per the CDNA5 ISA 16-bit A/B layout: lanes 0-15 hold K 0-7 & 16-23, lanes
// 16-31 hold K 8-15 & 24-31 -> two contiguous 16B chunks per lane.
__device__ __forceinline__ v16bf frag_ld(const unsigned short* p) {
    union { v16bf v; v4u u[2]; } r;
    r.u[0] = *(const v4u*)(p);        // K k0..k0+7
    r.u[1] = *(const v4u*)(p + 16);   // K k0+16..k0+23
    return r.v;
}

__device__ __forceinline__ v8f wmma_bf16(v16bf a, v16bf b, v8f c) {
    return __builtin_amdgcn_wmma_f32_16x16x32_bf16(
        false, a, false, b, (short)0, c, false, false);
}

// ---------------------------------------------------------------- cast kernel
__global__ void k_cvt_bf16(const float* __restrict__ x,
                           unsigned short* __restrict__ y, int n) {
    int i = (blockIdx.x * blockDim.x + threadIdx.x) * 4;
    if (i + 3 < n) {
        float4 v = *(const float4*)(x + i);
        y[i + 0] = f2bf(v.x);
        y[i + 1] = f2bf(v.y);
        y[i + 2] = f2bf(v.z);
        y[i + 3] = f2bf(v.w);
    }
}

// ---------------------------------------------------------------- router
// One block per token: logits, softmax, top-2, renormalized weights.
__global__ __launch_bounds__(256)
void k_router(const float* __restrict__ x, const float* __restrict__ Wr,
              const float* __restrict__ br, float* __restrict__ logits_out,
              int* __restrict__ topk_idx, float* __restrict__ topk_w) {
    int t = blockIdx.x;
    int tid = threadIdx.x;
    float acc[EE];
#pragma unroll
    for (int e = 0; e < EE; ++e) acc[e] = 0.f;

    const float* xr = x + (size_t)t * HH;
    for (int h = tid; h < HH; h += 256) {
        float xv = xr[h];
        const float* wr = Wr + (size_t)h * EE;
#pragma unroll
        for (int e = 0; e < EE; ++e) acc[e] += xv * wr[e];
    }

    __shared__ float red[256];
    __shared__ float logit[EE];
    for (int e = 0; e < EE; ++e) {
        red[tid] = acc[e];
        __syncthreads();
        for (int s = 128; s > 0; s >>= 1) {
            if (tid < s) red[tid] += red[tid + s];
            __syncthreads();
        }
        if (tid == 0) logit[e] = red[0] + br[e];
        __syncthreads();
    }

    if (tid == 0) {
        float m = logit[0];
#pragma unroll
        for (int e = 1; e < EE; ++e) m = fmaxf(m, logit[e]);
        float p[EE];
#pragma unroll
        for (int e = 0; e < EE; ++e) p[e] = __expf(logit[e] - m);
        int i0 = 0;
#pragma unroll
        for (int e = 1; e < EE; ++e) if (p[e] > p[i0]) i0 = e;
        int i1 = (i0 == 0) ? 1 : 0;
#pragma unroll
        for (int e = 0; e < EE; ++e)
            if (e != i0 && p[e] > p[i1]) i1 = e;
        float inv = 1.f / (p[i0] + p[i1]);
        topk_idx[t * 2 + 0] = i0;
        topk_idx[t * 2 + 1] = i1;
        topk_w[t * 2 + 0] = p[i0] * inv;
        topk_w[t * 2 + 1] = p[i1] * inv;
#pragma unroll
        for (int e = 0; e < EE; ++e)
            logits_out[(size_t)t * EE + e] = logit[e];
    }
}

// ---------------------------------------------------------------- compaction
// Single thread: deterministic per-expert token lists with 64-aligned bases.
// meta layout: [0..7]=counts [8..16]=slotBase[0..8] [17..25]=tileBase[0..8] [26]=totalTiles
__global__ void k_build(const int* __restrict__ topk_idx,
                        int* __restrict__ list, int* __restrict__ slot_of,
                        int* __restrict__ meta) {
    if (blockIdx.x != 0 || threadIdx.x != 0) return;
    int counts[EE];
    for (int e = 0; e < EE; ++e) counts[e] = 0;
    for (int t = 0; t < TT; ++t) {
        counts[topk_idx[t * 2 + 0]]++;
        counts[topk_idx[t * 2 + 1]]++;
    }
    int slotB[EE + 1], tileB[EE + 1];
    int sb = 0, tb = 0;
    for (int e = 0; e < EE; ++e) {
        slotB[e] = sb;
        tileB[e] = tb;
        int tl = (counts[e] + 63) >> 6;
        tb += tl;
        sb += tl * 64;
    }
    slotB[EE] = sb;
    tileB[EE] = tb;
    int fill[EE];
    for (int e = 0; e < EE; ++e) fill[e] = slotB[e];
    for (int t = 0; t < TT; ++t)
        for (int k = 0; k < KKTOP; ++k) {
            int e = topk_idx[t * 2 + k];
            int s = fill[e]++;
            list[s] = t;
            slot_of[t * 2 + k] = s;
        }
    for (int e = 0; e < EE; ++e)                 // padded tail rows -> token 0
        for (int s = fill[e]; s < slotB[e + 1]; ++s) list[s] = 0;
    for (int e = 0; e < EE; ++e) meta[e] = counts[e];
    for (int e = 0; e <= EE; ++e) { meta[8 + e] = slotB[e]; meta[17 + e] = tileB[e]; }
    meta[26] = tb;
}

// ---------------------------------------------------------------- gate+up GEMM
// Block tile: 64 tokens x 128 I-cols; 8 waves as 2x4; each wave = 2x2 WMMA
// tiles for gate AND up (8 v_wmma per 32-deep K step over H).
__global__ __launch_bounds__(256)
void k_gateup(const unsigned short* __restrict__ xbf,
              const float* __restrict__ Wg, const float* __restrict__ Wu,
              const int* __restrict__ list, const int* __restrict__ meta,
              unsigned short* __restrict__ act) {
    int b = blockIdx.x;
    if (b >= meta[26]) return;
    int ee = 0;
#pragma unroll
    for (int i = 0; i < EE; ++i)
        if (b >= meta[17 + i] && b < meta[18 + i]) ee = i;
    int tileIdx = b - meta[17 + ee];
    int slotBase = meta[8 + ee] + tileIdx * 64;
    const float* wg = Wg + (size_t)ee * HH * II;
    const float* wu = Wu + (size_t)ee * HH * II;

    int n0 = blockIdx.y * 128;
    int tid = threadIdx.x;
    int lane = tid & 31, wave = tid >> 5;
    int wm = (wave >> 2) * 32;   // wave row origin (0 / 32)
    int wn = (wave & 3) * 32;    // wave col origin (0..96)

    __shared__ __align__(16) unsigned short As[64 * 32];    //  4 KB
    __shared__ __align__(16) unsigned short Bgs[128 * 32];  //  8 KB (col-major: [n][k])
    __shared__ __align__(16) unsigned short Bus[128 * 32];  //  8 KB

    int arow = tid >> 2;             // 4 threads per A row
    int acol8 = (tid & 3) * 8;       // 8 bf16 = 16B per thread
    int token = list[slotBase + arow];
    const unsigned short* asrc = xbf + (size_t)token * HH + acol8;

    v8f accg[2][2] = {};
    v8f accu[2][2] = {};

    for (int k0 = 0; k0 < HH; k0 += 32) {
        __syncthreads();
        // A: gathered token rows, 16B vector copy.
        *(v4u*)(&As[arow * 32 + acol8]) = *(const v4u*)(asrc + k0);
        // B (gate & up): 32x128 fp32 each, convert -> bf16, store K-contiguous.
#pragma unroll
        for (int c = 0; c < 4; ++c) {
            int q = c * 256 + tid;        // 0..1023
            int kk = q >> 5;              // 0..31
            int n4 = (q & 31) * 4;        // 0..124
            size_t goff = (size_t)(k0 + kk) * II + n0 + n4;
            float4 vg = *(const float4*)(wg + goff);
            float4 vu = *(const float4*)(wu + goff);
            Bgs[(n4 + 0) * 32 + kk] = f2bf(vg.x);
            Bgs[(n4 + 1) * 32 + kk] = f2bf(vg.y);
            Bgs[(n4 + 2) * 32 + kk] = f2bf(vg.z);
            Bgs[(n4 + 3) * 32 + kk] = f2bf(vg.w);
            Bus[(n4 + 0) * 32 + kk] = f2bf(vu.x);
            Bus[(n4 + 1) * 32 + kk] = f2bf(vu.y);
            Bus[(n4 + 2) * 32 + kk] = f2bf(vu.z);
            Bus[(n4 + 3) * 32 + kk] = f2bf(vu.w);
        }
        __syncthreads();

        int koff = (lane >> 4) * 8;
        int rsel = lane & 15;
        v16bf a0  = frag_ld(&As[(wm + rsel) * 32 + koff]);
        v16bf a1  = frag_ld(&As[(wm + 16 + rsel) * 32 + koff]);
        v16bf bg0 = frag_ld(&Bgs[(wn + rsel) * 32 + koff]);
        v16bf bg1 = frag_ld(&Bgs[(wn + 16 + rsel) * 32 + koff]);
        v16bf bu0 = frag_ld(&Bus[(wn + rsel) * 32 + koff]);
        v16bf bu1 = frag_ld(&Bus[(wn + 16 + rsel) * 32 + koff]);

        accg[0][0] = wmma_bf16(a0, bg0, accg[0][0]);
        accg[0][1] = wmma_bf16(a0, bg1, accg[0][1]);
        accg[1][0] = wmma_bf16(a1, bg0, accg[1][0]);
        accg[1][1] = wmma_bf16(a1, bg1, accg[1][1]);
        accu[0][0] = wmma_bf16(a0, bu0, accu[0][0]);
        accu[0][1] = wmma_bf16(a0, bu1, accu[0][1]);
        accu[1][0] = wmma_bf16(a1, bu0, accu[1][0]);
        accu[1][1] = wmma_bf16(a1, bu1, accu[1][1]);
    }

    // Epilogue: act = silu(gate) * up, store bf16.
    int rofs = (lane >> 4) * 8;
    int csel = lane & 15;
#pragma unroll
    for (int im = 0; im < 2; ++im)
#pragma unroll
        for (int in = 0; in < 2; ++in)
#pragma unroll
            for (int r = 0; r < 8; ++r) {
                int row = wm + im * 16 + rofs + r;
                int col = n0 + wn + in * 16 + csel;
                float g = accg[im][in][r];
                float u = accu[im][in][r];
                float s = (g / (1.f + __expf(-g))) * u;
                act[(size_t)(slotBase + row) * II + col] = f2bf(s);
            }
}

// ---------------------------------------------------------------- down GEMM
// part[slot, 0..H) = act[slot, :] @ Wd[e]   (unscaled; combine applies weights)
__global__ __launch_bounds__(256)
void k_down(const unsigned short* __restrict__ act,
            const float* __restrict__ Wd, const int* __restrict__ meta,
            float* __restrict__ part) {
    int b = blockIdx.x;
    if (b >= meta[26]) return;
    int ee = 0;
#pragma unroll
    for (int i = 0; i < EE; ++i)
        if (b >= meta[17 + i] && b < meta[18 + i]) ee = i;
    int tileIdx = b - meta[17 + ee];
    int slotBase = meta[8 + ee] + tileIdx * 64;
    const float* wd = Wd + (size_t)ee * II * HH;

    int n0 = blockIdx.y * 128;
    int tid = threadIdx.x;
    int lane = tid & 31, wave = tid >> 5;
    int wm = (wave >> 2) * 32;
    int wn = (wave & 3) * 32;

    __shared__ __align__(16) unsigned short As[64 * 32];
    __shared__ __align__(16) unsigned short Bs[128 * 32];

    int arow = tid >> 2;
    int acol8 = (tid & 3) * 8;
    const unsigned short* asrc = act + (size_t)(slotBase + arow) * II + acol8;

    v8f acc[2][2] = {};

    for (int k0 = 0; k0 < II; k0 += 32) {
        __syncthreads();
        *(v4u*)(&As[arow * 32 + acol8]) = *(const v4u*)(asrc + k0);
#pragma unroll
        for (int c = 0; c < 4; ++c) {
            int q = c * 256 + tid;
            int kk = q >> 5;
            int n4 = (q & 31) * 4;
            float4 v = *(const float4*)(wd + (size_t)(k0 + kk) * HH + n0 + n4);
            Bs[(n4 + 0) * 32 + kk] = f2bf(v.x);
            Bs[(n4 + 1) * 32 + kk] = f2bf(v.y);
            Bs[(n4 + 2) * 32 + kk] = f2bf(v.z);
            Bs[(n4 + 3) * 32 + kk] = f2bf(v.w);
        }
        __syncthreads();

        int koff = (lane >> 4) * 8;
        int rsel = lane & 15;
        v16bf a0 = frag_ld(&As[(wm + rsel) * 32 + koff]);
        v16bf a1 = frag_ld(&As[(wm + 16 + rsel) * 32 + koff]);
        v16bf b0 = frag_ld(&Bs[(wn + rsel) * 32 + koff]);
        v16bf b1 = frag_ld(&Bs[(wn + 16 + rsel) * 32 + koff]);

        acc[0][0] = wmma_bf16(a0, b0, acc[0][0]);
        acc[0][1] = wmma_bf16(a0, b1, acc[0][1]);
        acc[1][0] = wmma_bf16(a1, b0, acc[1][0]);
        acc[1][1] = wmma_bf16(a1, b1, acc[1][1]);
    }

    int rofs = (lane >> 4) * 8;
    int csel = lane & 15;
#pragma unroll
    for (int im = 0; im < 2; ++im)
#pragma unroll
        for (int in = 0; in < 2; ++in)
#pragma unroll
            for (int r = 0; r < 8; ++r) {
                int row = wm + im * 16 + rofs + r;
                int col = n0 + wn + in * 16 + csel;
                part[(size_t)(slotBase + row) * HH + col] = acc[im][in][r];
            }
}

// ---------------------------------------------------------------- combine
__global__ __launch_bounds__(256)
void k_combine(const float* __restrict__ part, const int* __restrict__ slot_of,
               const float* __restrict__ tw, float* __restrict__ out) {
    int t = blockIdx.x;
    int tid = threadIdx.x;
    int s0 = slot_of[t * 2 + 0], s1 = slot_of[t * 2 + 1];
    float w0 = tw[t * 2 + 0], w1 = tw[t * 2 + 1];
    const float* p0 = part + (size_t)s0 * HH;
    const float* p1 = part + (size_t)s1 * HH;
    float* o = out + (size_t)t * HH;
#pragma unroll
    for (int j = 0; j < 2; ++j) {
        int h = tid * 8 + j * 4;
        float4 a = *(const float4*)(p0 + h);
        float4 b = *(const float4*)(p1 + h);
        float4 r;
        r.x = w0 * a.x + w1 * b.x;
        r.y = w0 * a.y + w1 * b.y;
        r.z = w0 * a.z + w1 * b.z;
        r.w = w0 * a.w + w1 * b.w;
        *(float4*)(o + h) = r;
    }
}

// ---------------------------------------------------------------- launch
extern "C" void kernel_launch(void* const* d_in, const int* in_sizes, int n_in,
                              void* d_out, int out_size, void* d_ws, size_t ws_size,
                              hipStream_t stream) {
    const float* x  = (const float*)d_in[0];  // [B,S,H]
    const float* Wr = (const float*)d_in[1];  // [H,E]
    const float* br = (const float*)d_in[2];  // [E]
    const float* Wg = (const float*)d_in[3];  // [E,H,I]
    const float* Wu = (const float*)d_in[4];  // [E,H,I]
    const float* Wd = (const float*)d_in[5];  // [E,I,H]
    float* out = (float*)d_out;               // [T*H] output, then [T*E] logits

    // Workspace carve (~363 MB total).
    char* w = (char*)d_ws;
    size_t off = 0;
    unsigned short* xbf = (unsigned short*)(w + off); off += (size_t)TT * HH * 2;
    int*   topk_idx = (int*)(w + off);   off += (size_t)TT * KKTOP * 4;
    float* topk_w   = (float*)(w + off); off += (size_t)TT * KKTOP * 4;
    int*   slot_of  = (int*)(w + off);   off += (size_t)TT * KKTOP * 4;
    int*   list     = (int*)(w + off);   off += (size_t)SLOT_CAP * 4;
    off = (off + 255) & ~(size_t)255;
    int*   meta     = (int*)(w + off);   off += 256;
    unsigned short* act = (unsigned short*)(w + off); off += (size_t)SLOT_CAP * II * 2;
    float* part     = (float*)(w + off); off += (size_t)SLOT_CAP * HH * 4;
    (void)ws_size; (void)in_sizes; (void)n_in; (void)out_size;

    k_cvt_bf16<<<(TT * HH) / 1024, 256, 0, stream>>>(x, xbf, TT * HH);
    k_router<<<TT, 256, 0, stream>>>(x, Wr, br, out + (size_t)TT * HH,
                                     topk_idx, topk_w);
    k_build<<<1, 32, 0, stream>>>(topk_idx, list, slot_of, meta);
    k_gateup<<<dim3(MAXTILES, II / 128), 256, 0, stream>>>(xbf, Wg, Wu, list,
                                                           meta, act);
    k_down<<<dim3(MAXTILES, HH / 128), 256, 0, stream>>>(act, Wd, meta, part);
    k_combine<<<TT, 256, 0, stream>>>(part, slot_of, topk_w, out);
}
